// NGMConvLayer_31061203484846
// MI455X (gfx1250) — matrix operand order repaired and checked by reading, not compiled, not run on hardware
//
#include <hip/hip_runtime.h>
#include <hip/hip_bf16.h>

typedef _Float16 half_t;
typedef half_t v16h __attribute__((ext_vector_type(16)));
typedef float  v8f  __attribute__((ext_vector_type(8)));
typedef float  v4f  __attribute__((ext_vector_type(4)));

#define BATCH 8
#define NNODE 512
#define NFEAT 32
#define BN    (BATCH * NNODE)      // 4096 rows
#define WELEMS ((size_t)BATCH * NNODE * NNODE * NFEAT)   // 67,108,864

// ---------------------------------------------------------------------------
// Kernel 1: the two tiny MLPs  (x @ W1 + b1).relu() @ W2 + b2 -> relu
// via v_wmma_f32_16x16x32_f16.  One wave computes one 16x32 output tile.
// grid = (32, 2) blocks of 256 threads (8 waves): 32*8 = 256 tiles * 16 rows
// = 4096 rows.  blockIdx.y: 0 -> n_func (x1), 1 -> n_self_func (s).
// ---------------------------------------------------------------------------
__global__ void __launch_bounds__(256)
ngm_mlp_wmma_kernel(const float* __restrict__ x,
                    const float* __restrict__ Wn1, const float* __restrict__ bn1,
                    const float* __restrict__ Wn2, const float* __restrict__ bn2,
                    const float* __restrict__ Ws1, const float* __restrict__ bs1,
                    const float* __restrict__ Ws2, const float* __restrict__ bs2,
                    float* __restrict__ x1out, float* __restrict__ sout)
{
    const int func = blockIdx.y;
    const float* W1 = func ? Ws1 : Wn1;
    const float* b1 = func ? bs1 : bn1;
    const float* W2 = func ? Ws2 : Wn2;
    const float* b2 = func ? bs2 : bn2;
    float* outp     = func ? sout : x1out;

    __shared__ float hbuf[8][16 * NFEAT];   // per-wave 16x32 relu(h) tile

    const int wv = threadIdx.x >> 5;        // wave in block
    const int l  = threadIdx.x & 31;        // lane
    const int g  = l >> 4;                  // lane half (K-group select)
    const int hl = l & 15;                  // lane within half (M or N index)
    const int m0 = (blockIdx.x * 8 + wv) * 16;   // first row of this tile

    // ---- layer 1 fragments -------------------------------------------------
    // A (16x32 f16): lane row = hl;  idx<8 -> K = 8g+idx ; idx>=8 -> K = 16+8g+(idx-8)
    v16h af;
#pragma unroll
    for (int idx = 0; idx < 16; ++idx) {
        int k = (idx < 8) ? (8 * g + idx) : (16 + 8 * g + (idx - 8));
        af[idx] = (half_t)x[(size_t)(m0 + hl) * NFEAT + k];
    }
    // B (32x16 f16), two column tiles: lane col = n*16+hl; idx<8 -> K = 8g+idx ; idx>=8 -> K = 16+8g+(idx-8)
    v16h bf0, bf1;
#pragma unroll
    for (int idx = 0; idx < 16; ++idx) {
        int k = (idx < 8) ? (8 * g + idx) : (16 + 8 * g + (idx - 8));
        bf0[idx] = (half_t)W1[k * NFEAT + hl];
        bf1[idx] = (half_t)W1[k * NFEAT + 16 + hl];
    }
    // C init = bias broadcast along rows
    v8f c0, c1;
    {
        float bb0 = b1[hl], bb1 = b1[16 + hl];
#pragma unroll
        for (int r = 0; r < 8; ++r) { c0[r] = bb0; c1[r] = bb1; }
    }
    c0 = __builtin_amdgcn_wmma_f32_16x16x32_f16(false, af, false, bf0, (short)0, c0, false, false);
    c1 = __builtin_amdgcn_wmma_f32_16x16x32_f16(false, af, false, bf1, (short)0, c1, false, false);

    // relu -> LDS (row-major 16x32) to re-layout D -> next A fragment
    float* hb = hbuf[wv];
#pragma unroll
    for (int r = 0; r < 8; ++r) {
        int m = r + 8 * g;
        hb[m * NFEAT + hl]      = fmaxf(c0[r], 0.0f);
        hb[m * NFEAT + 16 + hl] = fmaxf(c1[r], 0.0f);
    }
    __syncthreads();

    // ---- layer 2 -----------------------------------------------------------
    v16h af2;
#pragma unroll
    for (int idx = 0; idx < 16; ++idx) {
        int k = (idx < 8) ? (8 * g + idx) : (16 + 8 * g + (idx - 8));
        af2[idx] = (half_t)hb[hl * NFEAT + k];
    }
    v16h bf2a, bf2b;
#pragma unroll
    for (int idx = 0; idx < 16; ++idx) {
        int k = (idx < 8) ? (8 * g + idx) : (16 + 8 * g + (idx - 8));
        bf2a[idx] = (half_t)W2[k * NFEAT + hl];
        bf2b[idx] = (half_t)W2[k * NFEAT + 16 + hl];
    }
    v8f d0, d1;
    {
        float bb0 = b2[hl], bb1 = b2[16 + hl];
#pragma unroll
        for (int r = 0; r < 8; ++r) { d0[r] = bb0; d1[r] = bb1; }
    }
    d0 = __builtin_amdgcn_wmma_f32_16x16x32_f16(false, af2, false, bf2a, (short)0, d0, false, false);
    d1 = __builtin_amdgcn_wmma_f32_16x16x32_f16(false, af2, false, bf2b, (short)0, d1, false, false);

    // relu -> global (f32)
#pragma unroll
    for (int r = 0; r < 8; ++r) {
        int m = m0 + r + 8 * g;
        outp[(size_t)m * NFEAT + hl]      = fmaxf(d0[r], 0.0f);
        outp[(size_t)m * NFEAT + 16 + hl] = fmaxf(d1[r], 0.0f);
    }
}

// ---------------------------------------------------------------------------
// Kernel 2: fused W pass-through copy + L1 row norm of A + edge aggregation.
//   x2[b,i,f] = sum_j (A[b,i,j]/||A[b,i,:]||_1) * W[b,i,j,f] * x1[b,j,f] + s[b,i,f]
// One block per (b,i).  W streamed once with non-temporal hints (128b vectors).
// Thread t -> (jl = t>>3, f4 = t&7): 4 features starting at 4*f4.
// ---------------------------------------------------------------------------
__global__ void __launch_bounds__(256)
ngm_stream_kernel(const float* __restrict__ A,
                  const float* __restrict__ W,
                  const float* __restrict__ x1,
                  const float* __restrict__ sfx,
                  float* __restrict__ outW,
                  float* __restrict__ outX2)
{
    const int bi = blockIdx.x;            // b*512 + i
    const int t  = threadIdx.x;
    const int b  = bi >> 9;

    __shared__ float anrow[NNODE];        // normalized A row
    __shared__ float red[256];
    __shared__ v4f   part[256];           // [jl][f4] partial sums

    // ---- L1 norm of A[b,i,:] ----------------------------------------------
    const float* arow = A + (size_t)bi * NNODE;
    float a0 = arow[t];
    float a1 = arow[t + 256];
    red[t] = fabsf(a0) + fabsf(a1);
    __syncthreads();
#pragma unroll
    for (int s = 128; s > 0; s >>= 1) {
        if (t < s) red[t] += red[t + s];
        __syncthreads();
    }
    const float rn = 1.0f / fmaxf(red[0], 1e-12f);
    anrow[t]       = a0 * rn;
    anrow[t + 256] = a1 * rn;
    __syncthreads();

    // ---- stream W row: copy out + accumulate ------------------------------
    const int f4 = t & 7;                 // which 128b chunk of the 32 features
    const int jl = t >> 3;                // j within 32-wide group
    const v4f* Wv  = (const v4f*)W  + (size_t)bi * NNODE * 8;
    v4f*       Ov  = (v4f*)outW     + (size_t)bi * NNODE * 8;
    const v4f* X1v = (const v4f*)x1 + (size_t)b  * NNODE * 8;

    v4f acc = (v4f)0.0f;
#pragma unroll 4
    for (int it = 0; it < NNODE / 32; ++it) {
        const int j = it * 32 + jl;
        const size_t idx = (size_t)j * 8 + f4;
        v4f w = __builtin_nontemporal_load(&Wv[idx]);   // NT: no reuse
        __builtin_nontemporal_store(w, &Ov[idx]);       // NT: pass-through
        const float an = anrow[j];                      // LDS broadcast
        const v4f xv = X1v[idx];                        // L2-resident
        acc += (an * w) * xv;                           // componentwise fma
    }
    part[t] = acc;
    __syncthreads();

    // reduce over jl (32 groups), keeping f4 lanes independent
#pragma unroll
    for (int s = 16; s > 0; s >>= 1) {
        if (jl < s) part[t] += part[(jl + s) * 8 + f4];
        __syncthreads();
    }

    if (t < 8) {
        v4f r  = part[t];
        r += ((const v4f*)sfx)[(size_t)bi * 8 + t];
        ((v4f*)outX2)[(size_t)bi * 8 + t] = r;
    }
}

// ---------------------------------------------------------------------------
extern "C" void kernel_launch(void* const* d_in, const int* in_sizes, int n_in,
                              void* d_out, int out_size, void* d_ws, size_t ws_size,
                              hipStream_t stream)
{
    const float* A   = (const float*)d_in[0];
    const float* W   = (const float*)d_in[1];
    const float* x   = (const float*)d_in[2];
    const float* Wn1 = (const float*)d_in[3];
    const float* bn1 = (const float*)d_in[4];
    const float* Wn2 = (const float*)d_in[5];
    const float* bn2 = (const float*)d_in[6];
    const float* Ws1 = (const float*)d_in[7];
    const float* bs1 = (const float*)d_in[8];
    const float* Ws2 = (const float*)d_in[9];
    const float* bs2 = (const float*)d_in[10];

    float* outW  = (float*)d_out;
    float* outX2 = (float*)d_out + WELEMS;

    float* x1  = (float*)d_ws;                      // (B*N, F) f32 = 512 KB
    float* sfx = (float*)d_ws + (size_t)BN * NFEAT; // (B*N, F) f32 = 512 KB

    dim3 g1(32, 2);
    ngm_mlp_wmma_kernel<<<g1, 256, 0, stream>>>(x, Wn1, bn1, Wn2, bn2,
                                                Ws1, bs1, Ws2, bs2, x1, sfx);

    ngm_stream_kernel<<<BN, 256, 0, stream>>>(A, W, x1, sfx, outW, outX2);
}
